// MoELayer_89910845374649
// MI455X (gfx1250) — matrix operand order repaired and checked by reading, not compiled, not run on hardware
//
#include <hip/hip_runtime.h>
#include <hip/hip_bf16.h>

// ---------------- MoE: gate(top-2) + grouped bf16-WMMA FFN for gfx1250 ----------------
// T=65536 tokens, D=512, E=8, H=1024, TOP_K=2.

typedef __attribute__((ext_vector_type(16))) __bf16 v16bf;
typedef __attribute__((ext_vector_type(8)))  float  v8f;

constexpr int TKN = 65536;   // B*S
constexpr int DIM = 512;
constexpr int NEXP = 8;
constexpr int HID = 1024;
constexpr int MT  = 64;      // token rows per FFN block (4 m-tiles of 16)

// ---------------------------------------------------------------- WMMA helpers
__device__ __forceinline__ v8f wmma_bf16(v16bf a, v16bf b, v8f c) {
  return __builtin_amdgcn_wmma_f32_16x16x32_bf16(false, a, false, b, (short)0, c,
                                                 false, false);
}

// A fragment 16x32 bf16 (ISA 7.12.2): lanes0-15 M=lane, K in [kb,kb+8)+[kb+16,kb+24), kb=8*(lane>=16)
__device__ __forceinline__ v16bf load_frag_a(const __bf16* base, int ld, int k0, int lane) {
  const int row = lane & 15;
  const int kb  = (lane >> 4) << 3;
  const __bf16* p = base + (size_t)row * ld + k0 + kb;
  union { uint4 u[2]; v16bf v; } t;
  t.u[0] = *(const uint4*)(p);
  t.u[1] = *(const uint4*)(p + 16);
  return t.v;
}

// B fragment 32x16 bf16: lane&15 = column N, K in [kb,kb+8)+[kb+16,kb+24), kb=k0+8*(lane>=16).
// Weights pre-packed as [N][K] so this is two 16-byte loads.
__device__ __forceinline__ v16bf load_frag_b(const __bf16* wp, int ldk, int n0, int k0, int lane) {
  const int n  = n0 + (lane & 15);
  const int ks = k0 + ((lane >> 4) << 3);
  const __bf16* p = wp + (size_t)n * ldk + ks;
  union { uint4 u[2]; v16bf v; } t;
  t.u[0] = *(const uint4*)(p);
  t.u[1] = *(const uint4*)(p + 16);
  return t.v;
}

__device__ __forceinline__ float gelu_exact(float xv) {
  return 0.5f * xv * (1.0f + erff(xv * 0.70710678118f));
}

// ---------------------------------------------------------------- weight repack fp32 -> bf16 [e][N][K]
__global__ void moe_prep_kernel(const float* __restrict__ w1, const float* __restrict__ w2,
                                __bf16* __restrict__ w1p, __bf16* __restrict__ w2p) {
  const size_t N1 = (size_t)NEXP * DIM * HID;
  size_t id = (size_t)blockIdx.x * blockDim.x + threadIdx.x;
  if (id < N1) {
    // w1p[(e*H + n)*D + k] = w1[(e*D + k)*H + n]
    int e = (int)(id / ((size_t)DIM * HID));
    size_t r = id % ((size_t)DIM * HID);
    int n = (int)(r / DIM), k = (int)(r % DIM);
    w1p[id] = (__bf16)w1[((size_t)e * DIM + k) * HID + n];
  } else if (id < 2 * N1) {
    size_t id2 = id - N1;
    // w2p[(e*D + n)*H + k] = w2[(e*H + k)*D + n]
    int e = (int)(id2 / ((size_t)DIM * HID));
    size_t r = id2 % ((size_t)DIM * HID);
    int n = (int)(r / HID), k = (int)(r % HID);
    w2p[id2] = (__bf16)w2[((size_t)e * HID + k) * DIM + n];
  }
}

// ---------------------------------------------------------------- gating: softmax top-2 + scatter
__global__ __launch_bounds__(256) void moe_gate_kernel(
    const float* __restrict__ x, const float* __restrict__ gate_w,
    const float* __restrict__ gate_b, int* __restrict__ cnt,
    int* __restrict__ tok_list, float* __restrict__ wt_list) {
  __shared__ float gws[DIM * NEXP];      // 16 KB
  __shared__ float xbuf[256][33];        // padded rows -> conflict-free column reads
  const int tid = threadIdx.x;
  const int tb  = blockIdx.x * 256;

  for (int i = tid; i < DIM * NEXP; i += 256) gws[i] = gate_w[i];

  float acc[NEXP];
#pragma unroll
  for (int e = 0; e < NEXP; ++e) acc[e] = 0.0f;

  const int dsub = tid & 31, rw = tid >> 5;
  for (int dc = 0; dc < DIM; dc += 32) {
    __syncthreads();
#pragma unroll
    for (int rep = 0; rep < 8; ++rep) {            // coalesced 128B/wave loads of x
      int row = rw + rep * 8;
      xbuf[row][dsub] = x[(size_t)(tb + row) * DIM + dc + dsub];
    }
    __syncthreads();
    for (int dd = 0; dd < 32; ++dd) {
      float xv = xbuf[tid][dd];
#pragma unroll
      for (int e = 0; e < NEXP; ++e) acc[e] += xv * gws[(dc + dd) * NEXP + e];
    }
  }
#pragma unroll
  for (int e = 0; e < NEXP; ++e) acc[e] += gate_b[e];

  int i1 = 0; float s1 = acc[0];
#pragma unroll
  for (int e = 1; e < NEXP; ++e) if (acc[e] > s1) { s1 = acc[e]; i1 = e; }
  int i2 = (i1 == 0) ? 1 : 0; float s2 = acc[i2];
#pragma unroll
  for (int e = 0; e < NEXP; ++e) if (e != i1 && acc[e] > s2) { s2 = acc[e]; i2 = e; }

  // renormalized top-2 softmax weights: w1 = 1/(1+e^{s2-s1})
  float t  = __expf(s2 - s1);
  float wa = 1.0f / (1.0f + t);
  float wb = t * wa;

  const int tok = tb + tid;
  int p1 = atomicAdd(&cnt[i1], 1);
  tok_list[(size_t)i1 * TKN + p1] = tok; wt_list[(size_t)i1 * TKN + p1] = wa;
  int p2 = atomicAdd(&cnt[i2], 1);
  tok_list[(size_t)i2 * TKN + p2] = tok; wt_list[(size_t)i2 * TKN + p2] = wb;
}

// ---------------------------------------------------------------- grouped expert FFN (bf16 WMMA)
// 8 waves; each wave owns a disjoint slice of N-tiles and iterates all 4 M-tiles,
// so every weight (B) fragment is loaded once and feeds 4 WMMAs (A from LDS).
__global__ __launch_bounds__(256, 1) void moe_ffn_kernel(
    const float* __restrict__ x, const __bf16* __restrict__ w1p,
    const float* __restrict__ b1, const __bf16* __restrict__ w2p,
    const float* __restrict__ b2, const int* __restrict__ cnt,
    const int* __restrict__ tok_list, const float* __restrict__ wt_list,
    float* __restrict__ out) {
  const int e = blockIdx.y;
  const int nTok = cnt[e];
  const int rowBase = blockIdx.x * MT;
  if (rowBase >= nTok) return;

  extern __shared__ char smem[];                 // ~197 KB of the 320 KB WGP LDS
  __bf16* Xs  = (__bf16*)smem;                   // [MT][512]  bf16 : 64 KB
  __bf16* Hs  = Xs + MT * DIM;                   // [MT][1024] bf16 : 128 KB
  float*  scs = (float*)(Hs + MT * HID);         // [MT]
  int*    toks = (int*)(scs + MT);               // [MT]

  const int tid = threadIdx.x;
  if (tid < MT) {
    int r = rowBase + tid;
    if (r < nTok) {
      toks[tid] = tok_list[(size_t)e * TKN + r];
      scs[tid]  = wt_list[(size_t)e * TKN + r];
    } else {                                     // pad: valid row 0, zero weight
      toks[tid] = 0;
      scs[tid]  = 0.0f;
    }
  }
  __syncthreads();

  const int w = tid >> 5, lane = tid & 31;

  // gather 64 token rows fp32 -> bf16 LDS (coalesced float4 per wave)
  for (int rr = 0; rr < MT / 8; ++rr) {
    int r = w * 8 + rr;
    const float* xp = x + (size_t)toks[r] * DIM;
#pragma unroll
    for (int c = 0; c < 4; ++c) {
      float4 f = *(const float4*)(xp + c * 128 + lane * 4);
      union { __bf16 h[4]; uint2 u; } cv;
      cv.h[0] = (__bf16)f.x; cv.h[1] = (__bf16)f.y;
      cv.h[2] = (__bf16)f.z; cv.h[3] = (__bf16)f.w;
      *(uint2*)&Xs[(size_t)r * DIM + c * 128 + lane * 4] = cv.u;
    }
  }
  __syncthreads();

  // -------- stage A: H = gelu(X @ W1 + b1), kept in LDS as bf16 --------
  {
    const __bf16* w1e = w1p + (size_t)e * HID * DIM;
    for (int i = 0; i < (HID / 16) / 8; ++i) {   // 8 n-tiles per wave
      const int n0 = (w * 8 + i) * 16;
      __builtin_prefetch(w1e + (size_t)(n0 + 16) * DIM, 0, 1);
      v8f acc0 = {0.f,0.f,0.f,0.f,0.f,0.f,0.f,0.f};
      v8f acc1 = acc0, acc2 = acc0, acc3 = acc0;
#pragma unroll 2
      for (int kk = 0; kk < DIM / 32; ++kk) {
        const int k0 = kk * 32;
        v16bf b = load_frag_b(w1e, DIM, n0, k0, lane);
        acc0 = wmma_bf16(load_frag_a(Xs + 0 * 16 * DIM, DIM, k0, lane), b, acc0);
        acc1 = wmma_bf16(load_frag_a(Xs + 1 * 16 * DIM, DIM, k0, lane), b, acc1);
        acc2 = wmma_bf16(load_frag_a(Xs + 2 * 16 * DIM, DIM, k0, lane), b, acc2);
        acc3 = wmma_bf16(load_frag_a(Xs + 3 * 16 * DIM, DIM, k0, lane), b, acc3);
      }
      const int n = n0 + (lane & 15);
      const float bias = b1[(size_t)e * HID + n];
      const int mo = (lane >> 4) << 3;
      v8f* accs[4] = {&acc0, &acc1, &acc2, &acc3};
#pragma unroll
      for (int mt = 0; mt < 4; ++mt) {
        v8f a = *accs[mt];
#pragma unroll
        for (int v = 0; v < 8; ++v) {
          float g = gelu_exact(a[v] + bias);
          Hs[(size_t)(mt * 16 + mo + v) * HID + n] = (__bf16)g;
        }
      }
    }
  }
  __syncthreads();

  // -------- stage B: out += scale * (H @ W2 + b2) --------
  {
    const __bf16* w2e = w2p + (size_t)e * DIM * HID;
    for (int i = 0; i < (DIM / 16) / 8; ++i) {   // 4 n-tiles per wave
      const int n0 = (w * 4 + i) * 16;
      __builtin_prefetch(w2e + (size_t)(n0 + 16) * HID, 0, 1);
      v8f acc0 = {0.f,0.f,0.f,0.f,0.f,0.f,0.f,0.f};
      v8f acc1 = acc0, acc2 = acc0, acc3 = acc0;
#pragma unroll 2
      for (int kk = 0; kk < HID / 32; ++kk) {
        const int k0 = kk * 32;
        v16bf b = load_frag_b(w2e, HID, n0, k0, lane);
        acc0 = wmma_bf16(load_frag_a(Hs + 0 * 16 * HID, HID, k0, lane), b, acc0);
        acc1 = wmma_bf16(load_frag_a(Hs + 1 * 16 * HID, HID, k0, lane), b, acc1);
        acc2 = wmma_bf16(load_frag_a(Hs + 2 * 16 * HID, HID, k0, lane), b, acc2);
        acc3 = wmma_bf16(load_frag_a(Hs + 3 * 16 * HID, HID, k0, lane), b, acc3);
      }
      const int n = n0 + (lane & 15);
      const float bias = b2[(size_t)e * DIM + n];
      const int mo = (lane >> 4) << 3;
      v8f* accs[4] = {&acc0, &acc1, &acc2, &acc3};
#pragma unroll
      for (int mt = 0; mt < 4; ++mt) {
        v8f a = *accs[mt];
#pragma unroll
        for (int v = 0; v < 8; ++v) {
          const int m = mt * 16 + mo + v;
          const float s = scs[m];
          if (s != 0.0f) {  // exactly 2 contributions per out element -> deterministic
            atomicAdd(out + (size_t)toks[m] * DIM + n, s * (a[v] + bias));
          }
        }
      }
    }
  }
}

// ---------------------------------------------------------------- host launcher
extern "C" void kernel_launch(void* const* d_in, const int* in_sizes, int n_in,
                              void* d_out, int out_size, void* d_ws, size_t ws_size,
                              hipStream_t stream) {
  const float* x      = (const float*)d_in[0];
  const float* gate_w = (const float*)d_in[1];
  const float* gate_b = (const float*)d_in[2];
  const float* w1     = (const float*)d_in[3];
  const float* b1     = (const float*)d_in[4];
  const float* w2     = (const float*)d_in[5];
  const float* b2     = (const float*)d_in[6];
  float* out = (float*)d_out;
  char*  ws  = (char*)d_ws;

  // workspace layout (~21 MB)
  int*    cnt      = (int*)ws;                                   // 8 ints (256 B reserved)
  int*    tok_list = (int*)(ws + 256);                           // 8*T ints  = 2 MB
  float*  wt_list  = (float*)(ws + 256 + (size_t)NEXP * TKN * 4);// 8*T float = 2 MB
  __bf16* w1p      = (__bf16*)(ws + 256 + (size_t)NEXP * TKN * 8);
  __bf16* w2p      = w1p + (size_t)NEXP * HID * DIM;             // 8 MB each

  hipMemsetAsync(d_out, 0, (size_t)out_size * sizeof(float), stream);
  hipMemsetAsync(cnt, 0, 256, stream);

  const size_t nPrep = 2ull * NEXP * DIM * HID;
  moe_prep_kernel<<<(unsigned)((nPrep + 255) / 256), 256, 0, stream>>>(w1, w2, w1p, w2p);

  moe_gate_kernel<<<TKN / 256, 256, 0, stream>>>(x, gate_w, gate_b, cnt, tok_list, wt_list);

  constexpr size_t SMEM = (size_t)MT * DIM * 2 + (size_t)MT * HID * 2 + MT * 4 + MT * 4;
  hipFuncSetAttribute((const void*)moe_ffn_kernel,
                      hipFuncAttributeMaxDynamicSharedMemorySize, (int)SMEM);
  moe_ffn_kernel<<<dim3(TKN / MT, NEXP), 256, SMEM, stream>>>(
      x, w1p, b1, w2p, b2, cnt, tok_list, wt_list, out);
}
